// Effective_Draft_Decoder_49057116455199
// MI455X (gfx1250) — matrix operand order repaired and checked by reading, not compiled, not run on hardware
//
#include <hip/hip_runtime.h>
#include <math.h>

// ---------------- problem constants ----------------
#define HDIM   2048
#define NH     32
#define HD     64
#define VOC    32000
#define FF     5632
#define ENC    1536
#define LAB    512
#define PRED   8
#define SEQ    2048
#define INP    1024
#define MROWS  512          // rows 1536..2047 of the sequence (only these feed the head)

// ---------------- WMMA types ----------------
typedef __attribute__((ext_vector_type(16))) __bf16 v16bf;
typedef __attribute__((ext_vector_type(2)))  __bf16 v2bf;
typedef __attribute__((ext_vector_type(8)))  float  v8f;

union FragB16 { v16bf bf; unsigned short u[16]; };

// native fp32 -> bf16 (lets clang use the packed cvt datapath)
__device__ __forceinline__ unsigned short f2bf1(float f) {
  __bf16 h = (__bf16)f;
  unsigned short u;
  __builtin_memcpy(&u, &h, 2);
  return u;
}
__device__ __forceinline__ unsigned pack2bf(float x, float y) {
  v2bf h;
  h[0] = (__bf16)x;
  h[1] = (__bf16)y;
  unsigned u;
  __builtin_memcpy(&u, &h, 4);
  return u;
}
__device__ __forceinline__ v8f vzero8() {
  v8f z = {0.f,0.f,0.f,0.f,0.f,0.f,0.f,0.f};
  return z;
}

// CDNA5 async global->LDS copy (ASYNCcnt-tracked, 16B per lane)
__device__ __forceinline__ void async_b128(void* lds_dst, const void* gsrc) {
  unsigned loff = (unsigned)(unsigned long long)lds_dst;   // low 32 bits = LDS offset
  asm volatile("global_load_async_to_lds_b128 %0, %1, off"
               :: "v"(loff), "v"(gsrc) : "memory");
}
__device__ __forceinline__ void wait_async0() {
  asm volatile("s_wait_asynccnt 0" ::: "memory");
}

// A fragment: 16x32 bf16 tile, element(m,k) = p[m*stride + k]
__device__ __forceinline__ v16bf loadA16x32(const unsigned short* p, int stride) {
  int lane = threadIdx.x & 31;
  int row  = lane & 15;
  int half = lane >> 4;
  FragB16 f;
#pragma unroll
  for (int v = 0; v < 8; ++v) {
    int k = ((v & 3) << 1) + ((v >> 2) << 4) + (half << 3);
    f.u[2*v]   = p[row*stride + k];
    f.u[2*v+1] = p[row*stride + k + 1];
  }
  return f.bf;
}

// B fragment: 32x16 bf16 tile, element(k,n) = p[k*stride + n]
__device__ __forceinline__ v16bf loadB32x16(const unsigned short* p, int stride) {
  int lane = threadIdx.x & 31;
  int col  = lane & 15;
  int half = lane >> 4;
  FragB16 f;
#pragma unroll
  for (int v = 0; v < 8; ++v) {
    int k = ((v & 3) << 1) + ((v >> 2) << 4) + (half << 3);
    f.u[2*v]   = p[k*stride + col];
    f.u[2*v+1] = p[(k + 1)*stride + col];
  }
  return f.bf;
}

// B fragment from transposed view: element(k,n) = p[n*stride + k]
__device__ __forceinline__ v16bf loadBT32x16(const unsigned short* p, int stride) {
  int lane = threadIdx.x & 31;
  int col  = lane & 15;
  int half = lane >> 4;
  FragB16 f;
#pragma unroll
  for (int v = 0; v < 8; ++v) {
    int k = ((v & 3) << 1) + ((v >> 2) << 4) + (half << 3);
    f.u[2*v]   = p[col*stride + k];
    f.u[2*v+1] = p[col*stride + k + 1];
  }
  return f.bf;
}

#define WMMA_BF16(a,b,c) __builtin_amdgcn_wmma_f32_16x16x32_bf16(false,(a),false,(b),(short)0,(c),false,false)

// ---------------- build x = concat(encoder_out, emb[labels[:,INP:]]) ----------------
__global__ void k_build_x(const float4* __restrict__ enc, const int* __restrict__ labels,
                          const float4* __restrict__ emb, float4* __restrict__ x) {
  long i = (long)blockIdx.x * 256 + threadIdx.x;           // over SEQ*HDIM/4
  if (i >= (long)SEQ * (HDIM/4)) return;
  int r = (int)(i / (HDIM/4)), c = (int)(i % (HDIM/4));
  x[i] = (r < ENC) ? enc[i]
                   : emb[(long)labels[INP + (r - ENC)] * (HDIM/4) + c];
}

// ---------------- RMSNorm: fp32 in -> bf16 out ----------------
__global__ void k_rmsnorm(const float* __restrict__ x, const float* __restrict__ w,
                          unsigned short* __restrict__ out) {
  int row = blockIdx.x;
  const float4* xr = (const float4*)(x + (long)row * HDIM);
  const float4* w4 = (const float4*)w;
  __shared__ float red[256];
  float s = 0.f;
  for (int c = threadIdx.x; c < HDIM/4; c += 256) {
    float4 v = xr[c];
    s += v.x*v.x + v.y*v.y + v.z*v.z + v.w*v.w;
  }
  red[threadIdx.x] = s; __syncthreads();
  for (int o = 128; o > 0; o >>= 1) {
    if (threadIdx.x < o) red[threadIdx.x] += red[threadIdx.x + o];
    __syncthreads();
  }
  float scale = rsqrtf(red[0] / (float)HDIM + 1e-6f);
  uint2* o2 = (uint2*)(out + (long)row * HDIM);
  for (int c = threadIdx.x; c < HDIM/4; c += 256) {
    float4 v = xr[c], ww = w4[c];
    o2[c] = make_uint2(pack2bf(v.x*scale*ww.x, v.y*scale*ww.y),
                       pack2bf(v.z*scale*ww.z, v.w*scale*ww.w));
  }
}

// ---------------- WMMA GEMM: 128x128x64 tiles, 8 waves, double-buffered ----------------
// C(MxN) = A_bf16(MxK) * B_fp32(KxN)   (B converted to bf16 in LDS)
// modes: 0 = f32 out (+optional residual)   1 = bf16 out
//        2 = bf16 out + RoPE, pos = INP + m  3 = bf16 out + RoPE, pos = m<ENC ? m : m-LAB
#define GOUT_F32  0
#define GOUT_BF16 1
#define GROPE_Q   2
#define GROPE_K   3

#define GTM   128
#define GTN   128
#define KSTEP 64
#define BS_S  136    // 128 + 8 pad: 272B rows, 16B aligned, conflict-free frag reads

__global__ __launch_bounds__(256)
void k_gemm(const unsigned short* __restrict__ A, int lda,
            const float* __restrict__ B, int ldb,
            void* __restrict__ Cout, int ldc,
            const float* __restrict__ resid, int ldr,
            int K, int mode) {
  __shared__ unsigned short As[2][GTM * KSTEP];     // 2 x 16 KB
  __shared__ unsigned short Bs[2][KSTEP * BS_S];    // 2 x 17 KB
  int tid = threadIdx.x, wave = tid >> 5, lane = tid & 31;
  int wr = wave >> 1, wc = wave & 1;                // 4x2 wave grid
  int n0 = blockIdx.x * GTN, m0 = blockIdx.y * GTM;

  v8f acc[2][4];
#pragma unroll
  for (int fi = 0; fi < 2; ++fi)
#pragma unroll
    for (int j = 0; j < 4; ++j) acc[fi][j] = vzero8();

  auto loadTiles = [&](int kk, int buf) {
    // A tile: 128x64 bf16 = 1024 x 16B chunks, async DMA (4/thread)
#pragma unroll
    for (int ch = 0; ch < 4; ++ch) {
      int i = tid + ch * 256;
      int r = i >> 3, c8 = (i & 7) * 8;
      async_b128(&As[buf][r * KSTEP + c8], &A[(long)(m0 + r) * lda + kk + c8]);
    }
    // B tile: 64x128 fp32 -> bf16 (2048 float4 chunks, 8/thread)
#pragma unroll
    for (int ch = 0; ch < 8; ++ch) {
      int i = tid + ch * 256;
      int r = i >> 5, c4 = (i & 31) * 4;
      float4 v = *(const float4*)&B[(long)(kk + r) * ldb + n0 + c4];
      *(uint2*)&Bs[buf][r * BS_S + c4] =
          make_uint2(pack2bf(v.x, v.y), pack2bf(v.z, v.w));
    }
  };

  int cur = 0;
  loadTiles(0, 0);
  for (int kk = 0; kk < K; kk += KSTEP) {
    wait_async0();
    __syncthreads();
    if (kk + KSTEP < K) loadTiles(kk + KSTEP, cur ^ 1);
#pragma unroll
    for (int ks = 0; ks < KSTEP; ks += 32) {
      v16bf a0 = loadA16x32(&As[cur][(32 * wr +  0) * KSTEP + ks], KSTEP);
      v16bf a1 = loadA16x32(&As[cur][(32 * wr + 16) * KSTEP + ks], KSTEP);
      v16bf b0 = loadB32x16(&Bs[cur][ks * BS_S + 64 * wc +  0], BS_S);
      v16bf b1 = loadB32x16(&Bs[cur][ks * BS_S + 64 * wc + 16], BS_S);
      v16bf b2 = loadB32x16(&Bs[cur][ks * BS_S + 64 * wc + 32], BS_S);
      v16bf b3 = loadB32x16(&Bs[cur][ks * BS_S + 64 * wc + 48], BS_S);
      acc[0][0] = WMMA_BF16(a0, b0, acc[0][0]);
      acc[0][1] = WMMA_BF16(a0, b1, acc[0][1]);
      acc[0][2] = WMMA_BF16(a0, b2, acc[0][2]);
      acc[0][3] = WMMA_BF16(a0, b3, acc[0][3]);
      acc[1][0] = WMMA_BF16(a1, b0, acc[1][0]);
      acc[1][1] = WMMA_BF16(a1, b1, acc[1][1]);
      acc[1][2] = WMMA_BF16(a1, b2, acc[1][2]);
      acc[1][3] = WMMA_BF16(a1, b3, acc[1][3]);
    }
    cur ^= 1;
  }

  int half = lane >> 4, nl = lane & 15;

  if (mode == GOUT_F32) {
    float* C = (float*)Cout;
#pragma unroll
    for (int fi = 0; fi < 2; ++fi)
#pragma unroll
      for (int r = 0; r < 8; ++r) {
        int m = m0 + 32 * wr + 16 * fi + r + 8 * half;
#pragma unroll
        for (int j = 0; j < 4; ++j) {
          int n = n0 + 64 * wc + 16 * j + nl;
          float v = acc[fi][j][r];
          if (resid) v += resid[(long)m * ldr + n];
          C[(long)m * ldc + n] = v;
        }
      }
  } else if (mode == GOUT_BF16) {
    unsigned short* C = (unsigned short*)Cout;
#pragma unroll
    for (int fi = 0; fi < 2; ++fi)
#pragma unroll
      for (int r = 0; r < 8; ++r) {
        int m = m0 + 32 * wr + 16 * fi + r + 8 * half;
#pragma unroll
        for (int j = 0; j < 4; ++j)
          C[(long)m * ldc + n0 + 64 * wc + 16 * j + nl] = f2bf1(acc[fi][j][r]);
      }
  } else {
    // fused RoPE: heads are 64-wide; pairs (d, d+32) are acc[fi][j] / acc[fi][j+2]
    unsigned short* C = (unsigned short*)Cout;
#pragma unroll
    for (int fi = 0; fi < 2; ++fi)
#pragma unroll
      for (int r = 0; r < 8; ++r) {
        int m = m0 + 32 * wr + 16 * fi + r + 8 * half;
        int pos = (mode == GROPE_Q) ? (INP + m) : (m < ENC ? m : m - LAB);
#pragma unroll
        for (int j = 0; j < 2; ++j) {
          int d = 16 * j + nl;                                  // 0..31
          float inv = __expf(-(float)d * 0.28782313662425572f); // ln(10000)/32
          float sn, cs;
          __sincosf((float)pos * inv, &sn, &cs);
          float lo = acc[fi][j][r];
          float hi = acc[fi][j + 2][r];
          int n = n0 + 64 * wc + 16 * j + nl;
          C[(long)m * ldc + n]      = f2bf1(lo * cs - hi * sn);
          C[(long)m * ldc + n + 32] = f2bf1(hi * cs + lo * sn);
        }
      }
  }
}

// ---------------- flash attention: one (head, 16-q-row tile) per block ----------------
__global__ __launch_bounds__(128)
void k_attn(const unsigned short* __restrict__ qb, const unsigned short* __restrict__ kb,
            const unsigned short* __restrict__ vb, unsigned short* __restrict__ ob) {
  const int qtile = blockIdx.x;   // 0..31
  const int head  = blockIdx.y;   // 0..31
  const int KS_S  = HD + 8;

  __shared__ unsigned short Qs[16 * HD];
  __shared__ unsigned short Ks[64 * (HD + 8)];
  __shared__ unsigned short Vs[64 * (HD + 8)];
  __shared__ float          St[16 * 64];
  __shared__ unsigned short Pt[16 * 64];
  __shared__ float mrow[16], lrow[16], arow[16];

  int tid = threadIdx.x, wave = tid >> 5, lane = tid & 31;

  {   // Q tile: 128 x 16B chunks, one per thread
    int r = tid >> 3, c8 = (tid & 7) * 8;
    async_b128(&Qs[r * HD + c8],
               &qb[(long)(qtile * 16 + r) * HDIM + head * HD + c8]);
  }
  if (tid < 16) { mrow[tid] = -1e30f; lrow[tid] = 0.f; }

  v8f acc = vzero8();
  int qmax = ENC + qtile * 16 + 15;
  int nblocks = (qmax + 64) >> 6;

  for (int blk = 0; blk < nblocks; ++blk) {
    int k0 = blk * 64;
#pragma unroll
    for (int ch = 0; ch < 4; ++ch) {            // K,V tiles: 512 chunks each
      int i = tid + ch * 128;
      int r = i >> 3, c8 = (i & 7) * 8;
      async_b128(&Ks[r * KS_S + c8], &kb[(long)(k0 + r) * HDIM + head * HD + c8]);
      async_b128(&Vs[r * KS_S + c8], &vb[(long)(k0 + r) * HDIM + head * HD + c8]);
    }
    wait_async0();
    __syncthreads();

    // S = Q * K^T (each wave: 16 q rows x 16 keys)
    v8f sacc = vzero8();
#pragma unroll
    for (int kk = 0; kk < HD; kk += 32) {
      v16bf a = loadA16x32(&Qs[kk], HD);
      v16bf b = loadBT32x16(&Ks[(wave * 16) * KS_S + kk], KS_S);
      sacc = WMMA_BF16(a, b, sacc);
    }
    {
      int half = lane >> 4, nl = lane & 15;
#pragma unroll
      for (int r = 0; r < 8; ++r)
        St[(r + 8 * half) * 64 + wave * 16 + nl] = sacc[r] * 0.125f;
    }
    __syncthreads();

    // online softmax per q row (mask applied here)
    if (tid < 16) {
      int m  = tid;
      int qg = ENC + qtile * 16 + m;
      int bs = ENC + ((qg - ENC) / PRED) * PRED;
      int mlo = bs - LAB, mhi = bs;
      float mx = mrow[m];
      for (int j = 0; j < 64; ++j) {
        int kg = k0 + j;
        bool ok = (kg <= qg) && !((kg >= mlo) && (kg < mhi));
        float sv = ok ? St[m * 64 + j] : -1e30f;
        St[m * 64 + j] = sv;
        mx = fmaxf(mx, sv);
      }
      float al = __expf(mrow[m] - mx);
      float ls = lrow[m] * al;
      for (int j = 0; j < 64; ++j) {
        float p = __expf(St[m * 64 + j] - mx);
        Pt[m * 64 + j] = f2bf1(p);
        ls += p;
      }
      mrow[m] = mx; lrow[m] = ls; arow[m] = al;
    }
    __syncthreads();

    // O = O*alpha + P * V (each wave owns a 16-dim slice)
    {
      int half = lane >> 4;
#pragma unroll
      for (int r = 0; r < 8; ++r) acc[r] *= arow[r + 8 * half];
    }
#pragma unroll
    for (int kk = 0; kk < 64; kk += 32) {
      v16bf a = loadA16x32(&Pt[kk], 64);
      v16bf b = loadB32x16(&Vs[kk * KS_S + wave * 16], KS_S);
      acc = WMMA_BF16(a, b, acc);
    }
    __syncthreads();
  }

  int half = lane >> 4, nl = lane & 15;
#pragma unroll
  for (int r = 0; r < 8; ++r) {
    int m = r + 8 * half;
    float o = acc[r] / lrow[m];
    ob[(long)(qtile * 16 + m) * HDIM + head * HD + wave * 16 + nl] = f2bf1(o);
  }
}

// ---------------- silu(g) * u -> bf16 ----------------
__global__ void k_silu_gate(const float4* __restrict__ g, const float4* __restrict__ u,
                            uint2* __restrict__ t, long n4) {
  long i = (long)blockIdx.x * 256 + threadIdx.x;
  if (i >= n4) return;
  float4 gv = g[i], uv = u[i];
  float a = (gv.x / (1.f + __expf(-gv.x))) * uv.x;
  float b = (gv.y / (1.f + __expf(-gv.y))) * uv.y;
  float c = (gv.z / (1.f + __expf(-gv.z))) * uv.z;
  float d = (gv.w / (1.f + __expf(-gv.w))) * uv.w;
  t[i] = make_uint2(pack2bf(a, b), pack2bf(c, d));
}

// ---------------- loss ----------------
__global__ void k_zero(float* acc) { if (threadIdx.x < 2) acc[threadIdx.x] = 0.f; }

__global__ void k_loss(const float* __restrict__ logits, const float* __restrict__ llm,
                       const int* __restrict__ labels, float* __restrict__ acc) {
  int row = blockIdx.x;                     // 0..510
  const float*  lg  = logits + (long)row * VOC;
  const float4* lg4 = (const float4*)lg;
  const float4* lp4 = (const float4*)(llm + (long)row * VOC);
  __shared__ float red[256];
  int tid = threadIdx.x;

  float m1 = -3e38f, m2 = -3e38f;
  for (int j = tid; j < VOC/4; j += 256) {
    float4 a = lg4[j], b = lp4[j];
    m1 = fmaxf(m1, fmaxf(fmaxf(a.x, a.y), fmaxf(a.z, a.w)));
    m2 = fmaxf(m2, fmaxf(fmaxf(b.x, b.y), fmaxf(b.z, b.w)));
  }
  red[tid] = m1; __syncthreads();
  for (int o = 128; o > 0; o >>= 1) { if (tid < o) red[tid] = fmaxf(red[tid], red[tid + o]); __syncthreads(); }
  m1 = red[0]; __syncthreads();
  red[tid] = m2; __syncthreads();
  for (int o = 128; o > 0; o >>= 1) { if (tid < o) red[tid] = fmaxf(red[tid], red[tid + o]); __syncthreads(); }
  m2 = red[0]; __syncthreads();

  float z1 = 0.f, z2 = 0.f;
  for (int j = tid; j < VOC/4; j += 256) {
    float4 a = lg4[j], b = lp4[j];
    z1 += __expf(a.x - m1) + __expf(a.y - m1) + __expf(a.z - m1) + __expf(a.w - m1);
    z2 += __expf(b.x - m2) + __expf(b.y - m2) + __expf(b.z - m2) + __expf(b.w - m2);
  }
  red[tid] = z1; __syncthreads();
  for (int o = 128; o > 0; o >>= 1) { if (tid < o) red[tid] += red[tid + o]; __syncthreads(); }
  z1 = red[0]; __syncthreads();
  red[tid] = z2; __syncthreads();
  for (int o = 128; o > 0; o >>= 1) { if (tid < o) red[tid] += red[tid + o]; __syncthreads(); }
  z2 = red[0]; __syncthreads();

  float lz1 = logf(z1), lz2 = logf(z2);
  float kl = 0.f;
  for (int j = tid; j < VOC/4; j += 256) {
    float4 a = lg4[j], b = lp4[j];
    float lq0 = a.x - m1 - lz1, lp0 = b.x - m2 - lz2;
    float lq1 = a.y - m1 - lz1, lp1 = b.y - m2 - lz2;
    float lq2 = a.z - m1 - lz1, lp2 = b.z - m2 - lz2;
    float lq3 = a.w - m1 - lz1, lp3 = b.w - m2 - lz2;
    kl += __expf(lp0) * (lp0 - lq0) + __expf(lp1) * (lp1 - lq1)
        + __expf(lp2) * (lp2 - lq2) + __expf(lp3) * (lp3 - lq3);
  }
  red[tid] = kl; __syncthreads();
  for (int o = 128; o > 0; o >>= 1) { if (tid < o) red[tid] += red[tid + o]; __syncthreads(); }
  if (tid == 0) {
    atomicAdd(&acc[1], red[0]);
    int lab = labels[INP + 1 + row];        // labels[0, 1025+row]
    atomicAdd(&acc[0], -(lg[lab] - m1 - lz1));
  }
}

__global__ void k_finalize(const float* __restrict__ acc, float* __restrict__ out) {
  if (threadIdx.x == 0) {
    out[0] = acc[0] / 511.f;   // ce
    out[1] = acc[1] / 511.f;   // kl
  }
}

// ---------------- launch ----------------
extern "C" void kernel_launch(void* const* d_in, const int* in_sizes, int n_in,
                              void* d_out, int out_size, void* d_ws, size_t ws_size,
                              hipStream_t stream) {
  (void)in_sizes; (void)n_in; (void)out_size; (void)ws_size;
  const float* enc    = (const float*)d_in[0];
  const int*   labels = (const int*)  d_in[1];
  const float* llm    = (const float*)d_in[2];
  const float* emb    = (const float*)d_in[3];
  const float* Wq     = (const float*)d_in[4];
  const float* Wk     = (const float*)d_in[5];
  const float* Wv     = (const float*)d_in[6];
  const float* Wo     = (const float*)d_in[7];
  const float* Wg     = (const float*)d_in[8];
  const float* Wu     = (const float*)d_in[9];
  const float* Wd     = (const float*)d_in[10];
  const float* wln1   = (const float*)d_in[11];
  const float* wln2   = (const float*)d_in[12];
  const float* wnorm  = (const float*)d_in[13];
  const float* Wlm    = (const float*)d_in[14];

  char* wp = (char*)d_ws;
  auto alloc = [&](size_t bytes) -> void* {
    void* p = (void*)wp;
    wp += (bytes + 255) & ~(size_t)255;
    return p;
  };
  float*          x      = (float*)         alloc((size_t)SEQ  * HDIM * 4);
  unsigned short* h1     = (unsigned short*)alloc((size_t)SEQ  * HDIM * 2);
  unsigned short* qbuf   = (unsigned short*)alloc((size_t)MROWS* HDIM * 2);
  unsigned short* kbuf   = (unsigned short*)alloc((size_t)SEQ  * HDIM * 2);
  unsigned short* vbuf   = (unsigned short*)alloc((size_t)SEQ  * HDIM * 2);
  unsigned short* obuf   = (unsigned short*)alloc((size_t)MROWS* HDIM * 2);
  float*          x2     = (float*)         alloc((size_t)MROWS* HDIM * 4);
  unsigned short* h2     = (unsigned short*)alloc((size_t)MROWS* HDIM * 2);
  float*          g      = (float*)         alloc((size_t)MROWS* FF   * 4);
  float*          u      = (float*)         alloc((size_t)MROWS* FF   * 4);
  unsigned short* t      = (unsigned short*)alloc((size_t)MROWS* FF   * 2);
  float*          x3     = (float*)         alloc((size_t)MROWS* HDIM * 4);
  unsigned short* h3     = (unsigned short*)alloc((size_t)MROWS* HDIM * 2);
  float*          logits = (float*)         alloc((size_t)MROWS* VOC  * 4);
  float*          acc    = (float*)         alloc(256);

  dim3 blk(256);

  k_build_x<<<(SEQ * (HDIM/4) + 255) / 256, 256, 0, stream>>>(
      (const float4*)enc, labels, (const float4*)emb, (float4*)x);
  k_rmsnorm<<<SEQ, 256, 0, stream>>>(x, wln1, h1);

  // q (last 512 rows, RoPE pos = 1024+m), k (all rows, RoPE), v (all rows)
  k_gemm<<<dim3(HDIM/GTN, MROWS/GTM), blk, 0, stream>>>(h1 + (size_t)ENC*HDIM, HDIM, Wq, HDIM,
                                                        qbuf, HDIM, nullptr, 0, HDIM, GROPE_Q);
  k_gemm<<<dim3(HDIM/GTN, SEQ/GTM),   blk, 0, stream>>>(h1, HDIM, Wk, HDIM,
                                                        kbuf, HDIM, nullptr, 0, HDIM, GROPE_K);
  k_gemm<<<dim3(HDIM/GTN, SEQ/GTM),   blk, 0, stream>>>(h1, HDIM, Wv, HDIM,
                                                        vbuf, HDIM, nullptr, 0, HDIM, GOUT_BF16);

  k_attn<<<dim3(MROWS/16, NH), dim3(128), 0, stream>>>(qbuf, kbuf, vbuf, obuf);

  // x2 = x[1536:] + o @ Wo
  k_gemm<<<dim3(HDIM/GTN, MROWS/GTM), blk, 0, stream>>>(obuf, HDIM, Wo, HDIM,
                                                        x2, HDIM, x + (size_t)ENC*HDIM, HDIM,
                                                        HDIM, GOUT_F32);
  k_rmsnorm<<<MROWS, 256, 0, stream>>>(x2, wln2, h2);

  k_gemm<<<dim3(FF/GTN, MROWS/GTM), blk, 0, stream>>>(h2, HDIM, Wg, FF, g, FF, nullptr, 0, HDIM, GOUT_F32);
  k_gemm<<<dim3(FF/GTN, MROWS/GTM), blk, 0, stream>>>(h2, HDIM, Wu, FF, u, FF, nullptr, 0, HDIM, GOUT_F32);
  k_silu_gate<<<(int)(((size_t)MROWS*FF/4 + 255) / 256), 256, 0, stream>>>(
      (const float4*)g, (const float4*)u, (uint2*)t, (long)MROWS*FF/4);
  k_gemm<<<dim3(HDIM/GTN, MROWS/GTM), blk, 0, stream>>>(t, FF, Wd, HDIM, x3, HDIM, x2, HDIM, FF, GOUT_F32);

  k_rmsnorm<<<MROWS, 256, 0, stream>>>(x3, wnorm, h3);
  k_gemm<<<dim3(VOC/GTN, MROWS/GTM), blk, 0, stream>>>(h3, HDIM, Wlm, VOC, logits, VOC, nullptr, 0, HDIM, GOUT_F32);

  k_zero<<<1, 32, 0, stream>>>(acc);
  k_loss<<<LAB - 1, 256, 0, stream>>>(logits, llm, labels, acc);
  k_finalize<<<1, 32, 0, stream>>>(acc, (float*)d_out);
}
